// _LBlock_5961414607313
// MI455X (gfx1250) — matrix operand order repaired and checked by reading, not compiled, not run on hardware
//
#include <hip/hip_runtime.h>
#include <hip/hip_bf16.h>
#include <math.h>

// ---------------- problem constants ----------------
#define NN   131072      // nodes
#define DD   128         // hidden dim
#define HH   4           // heads
#define HDIM 32          // head dim
#define TT   3           // edge types
#define EE   262144      // edges per type
#define BB   512         // graphs
#define SLOPE 0.2f
#define EPS   1e-6f

typedef __bf16 bf16_t;
typedef __attribute__((ext_vector_type(16))) __bf16 v16bf;
typedef __attribute__((ext_vector_type(8)))  __bf16 v8bf;
typedef __attribute__((ext_vector_type(4)))  __bf16 v4bf;
typedef __attribute__((ext_vector_type(8)))  float  v8f;

#define LDS_STRIDE 136   // bf16 elements per row (pad vs 128 to avoid bank conflicts; mult of 8)
#define GEMM_LDS_BYTES (2 * 128 * LDS_STRIDE * 2)  // X tile + W^T tile, bf16

// ---------------- helpers ----------------
__device__ __forceinline__ float gelu_exact(float x) {
    return 0.5f * x * (1.0f + erff(x * 0.70710678118654752440f));
}
__device__ __forceinline__ float lrelu(float x) { return x > 0.0f ? x : SLOPE * x; }

// order-preserving float<->uint encoding for atomicMax-based segment max
__device__ __forceinline__ unsigned encf(float f) {
    unsigned u = __float_as_uint(f);
    return (u & 0x80000000u) ? ~u : (u | 0x80000000u);
}
__device__ __forceinline__ float decf(unsigned u) {
    return (u & 0x80000000u) ? __uint_as_float(u ^ 0x80000000u) : __uint_as_float(~u);
}

__device__ __forceinline__ float wave_sum32(float v) {
#pragma unroll
    for (int m = 16; m > 0; m >>= 1) v += __shfl_xor(v, m, 32);
    return v;
}

// ---------------- WMMA GEMM: Y[M x 128] = op(X[M x 128]) @ W[128 x 128] (+bias) ----------------
// wmode: 0 = dense W (row-major [k][c]); 1 = block-diagonal from rel[H][32][32]
// ingelu: apply exact GELU to X while staging; accum: Y += result instead of Y = result
__global__ void gemm128_wmma(const float* __restrict__ X, const float* __restrict__ W,
                             const float* __restrict__ bias, float* __restrict__ Y,
                             int wmode, int ingelu, int accum)
{
    extern __shared__ char smem[];
    bf16_t* ldsX = (bf16_t*)smem;                              // [128][LDS_STRIDE]
    bf16_t* ldsW = (bf16_t*)(smem + 128 * LDS_STRIDE * 2);     // transposed: [col][LDS_STRIDE]

    const int tid = threadIdx.x;
    const long rowBase = (long)blockIdx.x * 128;

    // stage W (transposed) as bf16
    if (wmode == 0) {
        const float4* W4 = (const float4*)W;
        for (int i = tid; i < 128 * 32; i += 256) {     // (k, c4)
            int k = i >> 5, c4 = i & 31;
            float4 w = W4[k * 32 + c4];
            int c = c4 * 4;
            ldsW[(c + 0) * LDS_STRIDE + k] = (bf16_t)w.x;
            ldsW[(c + 1) * LDS_STRIDE + k] = (bf16_t)w.y;
            ldsW[(c + 2) * LDS_STRIDE + k] = (bf16_t)w.z;
            ldsW[(c + 3) * LDS_STRIDE + k] = (bf16_t)w.w;
        }
    } else {
        for (int i = tid; i < 128 * 128; i += 256) {
            int k = i >> 7, c = i & 127;
            int hk = k >> 5, hc = c >> 5;
            float w = (hk == hc) ? W[(hk * 32 + (k & 31)) * 32 + (c & 31)] : 0.0f;
            ldsW[c * LDS_STRIDE + k] = (bf16_t)w;
        }
    }
    // stage X tile as bf16 (optionally through GELU)
    {
        const float4* X4 = (const float4*)(X + rowBase * DD);
        for (int i = tid; i < 128 * 32; i += 256) {     // (r, c4)
            int r = i >> 5, c4 = i & 31;
            float4 v = X4[r * 32 + c4];
            if (ingelu) {
                v.x = gelu_exact(v.x); v.y = gelu_exact(v.y);
                v.z = gelu_exact(v.z); v.w = gelu_exact(v.w);
            }
            v4bf bv;
            bv[0] = (bf16_t)v.x; bv[1] = (bf16_t)v.y;
            bv[2] = (bf16_t)v.z; bv[3] = (bf16_t)v.w;
            *(v4bf*)(ldsX + r * LDS_STRIDE + c4 * 4) = bv;   // 8B aligned (LDS_STRIDE%8==0)
        }
    }
    __syncthreads();

    const int wave = tid >> 5;
    const int lane = tid & 31;
    const int hl   = lane >> 4;   // half-wave select
    const int r    = lane & 15;
    const int m0   = wave * 16;   // this wave's 16-row strip

    v8f acc[8] = {};

#pragma unroll
    for (int kb = 0; kb < 128; kb += 32) {
        // A fragment (16x32 bf16): lane<16 holds K {kb..kb+7, kb+16..kb+23} of row m0+r,
        // lane>=16 holds K {kb+8..kb+15, kb+24..kb+31}
        const bf16_t* xa = ldsX + (m0 + r) * LDS_STRIDE + kb + hl * 8;
        v8bf alo = *(const v8bf*)xa;
        v8bf ahi = *(const v8bf*)(xa + 16);
        v16bf A = __builtin_shufflevector(alo, ahi, 0,1,2,3,4,5,6,7,8,9,10,11,12,13,14,15);

        // load ALL 8 B fragments into distinct registers first so the WMMAs below can
        // overlap with outstanding ds_loads (graduated s_wait_dscnt instead of 0-drain)
        v16bf Bm[8];
#pragma unroll
        for (int t = 0; t < 8; ++t) {
            const bf16_t* wb = ldsW + (t * 16 + r) * LDS_STRIDE + kb + hl * 8;
            v8bf blo = *(const v8bf*)wb;
            v8bf bhi = *(const v8bf*)(wb + 16);
            Bm[t] = __builtin_shufflevector(blo, bhi, 0,1,2,3,4,5,6,7,8,9,10,11,12,13,14,15);
        }
#pragma unroll
        for (int t = 0; t < 8; ++t) {
            acc[t] = __builtin_amdgcn_wmma_f32_16x16x32_bf16(
                false, A, false, Bm[t], (short)0, acc[t], false, false);
        }
    }

    // epilogue: C/D layout -> element v of acc = D[m0 + v + 8*hl][t*16 + r]
#pragma unroll
    for (int t = 0; t < 8; ++t) {
        int col = t * 16 + r;
        float bv = bias ? bias[col] : 0.0f;
#pragma unroll
        for (int v = 0; v < 8; ++v) {
            long idx = (rowBase + m0 + v + hl * 8) * DD + col;
            float rv = acc[t][v] + bv;
            if (accum) rv += Y[idx];
            Y[idx] = rv;
        }
    }
}

// ---------------- small utility kernels ----------------
__global__ void fill_u32(unsigned* p, unsigned v, long n) {
    long i = (long)blockIdx.x * blockDim.x + threadIdx.x;
    if (i < n) p[i] = v;
}

// asrc[n,h] = sum_d h[n,h*32+d]*a_src[h,d];  adst likewise
__global__ void gat_attn_coef(const float* __restrict__ h, const float* __restrict__ a_src,
                              const float* __restrict__ a_dst,
                              float* __restrict__ asrc, float* __restrict__ adst)
{
    int i = blockIdx.x * blockDim.x + threadIdx.x;     // [0, N*H)
    if (i >= NN * HH) return;
    int n = i >> 2, hh = i & 3;
    const float* hp = h + (long)n * DD + hh * HDIM;
    const float* ap = a_src + hh * HDIM;
    const float* bp = a_dst + hh * HDIM;
    float s1 = 0.0f, s2 = 0.0f;
#pragma unroll
    for (int d = 0; d < HDIM; ++d) { float v = hp[d]; s1 += v * ap[d]; s2 += v * bp[d]; }
    asrc[i] = s1; adst[i] = s2;
}

// pass 1: segment max of LeakyReLU(asrc[src]+adst[dst]); src==dst==e when ptrs are null (self loops)
__global__ void gat_edge_max(const int* src, const int* dst, const float* __restrict__ asrc,
                             const float* __restrict__ adst, unsigned* __restrict__ menc, int cnt)
{
    int i = blockIdx.x * blockDim.x + threadIdx.x;
    if (i >= cnt * HH) return;
    int e = i >> 2, hh = i & 3;
    int s = src ? src[e] : e, d = dst ? dst[e] : e;
    float sc = lrelu(asrc[s * HH + hh] + adst[d * HH + hh]);
    atomicMax(&menc[d * HH + hh], encf(sc));
}

// pass 2: segment sum of exp(score - max)
__global__ void gat_edge_sum(const int* src, const int* dst, const float* __restrict__ asrc,
                             const float* __restrict__ adst, const unsigned* __restrict__ menc,
                             float* __restrict__ ssum, int cnt)
{
    int i = blockIdx.x * blockDim.x + threadIdx.x;
    if (i >= cnt * HH) return;
    int e = i >> 2, hh = i & 3;
    int s = src ? src[e] : e, d = dst ? dst[e] : e;
    float sc = lrelu(asrc[s * HH + hh] + adst[d * HH + hh]);
    float ex = __expf(sc - decf(menc[d * HH + hh]));
    atomicAdd(&ssum[d * HH + hh], ex);
}

// pass 3: out[dst] += alpha * h[src]   (one wave per edge; lane = head-dim)
__global__ void gat_edge_agg(const int* src, const int* dst, const float* __restrict__ asrc,
                             const float* __restrict__ adst, const unsigned* __restrict__ menc,
                             const float* __restrict__ ssum, const float* __restrict__ hbuf,
                             float* __restrict__ out, int cnt)
{
    int w = blockIdx.x * (blockDim.x >> 5) + (threadIdx.x >> 5);
    if (w >= cnt) return;
    int lane = threadIdx.x & 31;
    int s = src ? src[w] : w, d = dst ? dst[w] : w;
#pragma unroll
    for (int hh = 0; hh < HH; ++hh) {
        float sc = lrelu(asrc[s * HH + hh] + adst[d * HH + hh]);
        float alpha = __expf(sc - decf(menc[d * HH + hh])) / ssum[d * HH + hh];
        float v = hbuf[(long)s * DD + hh * HDIM + lane];
        atomicAdd(&out[(long)d * DD + hh * HDIM + lane], alpha * v);
    }
}

// residual: xout = LeakyReLU(xold + rmsnorm(xnew + sum_t b[t], scale)); one wave per node
__global__ void res_kernel(const float* __restrict__ xold, const float* __restrict__ xnew,
                           const float* __restrict__ b3, const float* __restrict__ scale,
                           float* __restrict__ xout)
{
    int node = blockIdx.x * (blockDim.x >> 5) + (threadIdx.x >> 5);
    if (node >= NN) return;
    int lane = threadIdx.x & 31;
    const long base = (long)node * DD;
    float v[4], ss = 0.0f;
#pragma unroll
    for (int j = 0; j < 4; ++j) {
        int c = lane + 32 * j;
        float bb = b3 ? (b3[c] + b3[DD + c] + b3[2 * DD + c]) : 0.0f;
        v[j] = xnew[base + c] + bb;
        ss += v[j] * v[j];
    }
    ss = wave_sum32(ss);
    float rr = rsqrtf(ss * (1.0f / DD) + EPS);
#pragma unroll
    for (int j = 0; j < 4; ++j) {
        int c = lane + 32 * j;
        xout[base + c] = lrelu(xold[base + c] + scale[c] * v[j] * rr);
    }
}

// HGT pass 1: score = (q[dst].kt[src]) * rel_p / sqrt(HD); cache score + segment max. wave/edge.
__global__ void hgt_score_max(const int* __restrict__ src, const int* __restrict__ dst,
                              const float* __restrict__ q, const float* __restrict__ kt,
                              const float* __restrict__ relp, float* __restrict__ scorebuf,
                              unsigned* __restrict__ menc)
{
    int w = blockIdx.x * (blockDim.x >> 5) + (threadIdx.x >> 5);
    if (w >= EE) return;
    int lane = threadIdx.x & 31;
    int s = src[w], d = dst[w];
#pragma unroll
    for (int hh = 0; hh < HH; ++hh) {
        float p = q[(long)d * DD + hh * HDIM + lane] * kt[(long)s * DD + hh * HDIM + lane];
        p = wave_sum32(p);
        float sc = p * relp[hh] * 0.17677669529663687f;   // 1/sqrt(32)
        if (lane == 0) {
            scorebuf[(long)w * HH + hh] = sc;
            atomicMax(&menc[d * HH + hh], encf(sc));
        }
    }
}

// HGT pass 2: segment sum of exp over all T*E edges (scores cached)
__global__ void hgt_sum(const float* __restrict__ score, const int* __restrict__ ei,
                        const unsigned* __restrict__ menc, float* __restrict__ ssum)
{
    long i = (long)blockIdx.x * blockDim.x + threadIdx.x;     // [0, T*E*H)
    if (i >= (long)TT * EE * HH) return;
    long te = i >> 2; int hh = (int)(i & 3);
    int t = (int)(te >> 18);            // E = 2^18
    int e = (int)(te & (EE - 1));
    int d = ei[t * 2 * EE + EE + e];
    float ex = __expf(score[i] - decf(menc[d * HH + hh]));
    atomicAdd(&ssum[d * HH + hh], ex);
}

// HGT pass 3: out[dst] += alpha * vt[src]; one wave per edge (per type)
__global__ void hgt_agg(const int* __restrict__ src, const int* __restrict__ dst,
                        const float* __restrict__ score, const unsigned* __restrict__ menc,
                        const float* __restrict__ ssum, const float* __restrict__ vt,
                        float* __restrict__ out)
{
    int w = blockIdx.x * (blockDim.x >> 5) + (threadIdx.x >> 5);
    if (w >= EE) return;
    int lane = threadIdx.x & 31;
    int s = src[w], d = dst[w];
#pragma unroll
    for (int hh = 0; hh < HH; ++hh) {
        float sc = score[(long)w * HH + hh];
        float alpha = __expf(sc - decf(menc[d * HH + hh])) / ssum[d * HH + hh];
        atomicAdd(&out[(long)d * DD + hh * HDIM + lane],
                  alpha * vt[(long)s * DD + hh * HDIM + lane]);
    }
}

// skip-gate + residual norm: x3 = LeakyReLU(x2 + rmsnorm(g*y + (1-g)*x2, scale))
__global__ void hgt_res(const float* __restrict__ x2, const float* __restrict__ y,
                        const float* __restrict__ skip, const float* __restrict__ scale,
                        float* __restrict__ xout)
{
    int node = blockIdx.x * (blockDim.x >> 5) + (threadIdx.x >> 5);
    if (node >= NN) return;
    int lane = threadIdx.x & 31;
    const long base = (long)node * DD;
    float g = 1.0f / (1.0f + __expf(-skip[0]));
    float v[4], xo[4], ss = 0.0f;
#pragma unroll
    for (int j = 0; j < 4; ++j) {
        int c = lane + 32 * j;
        xo[j] = x2[base + c];
        v[j] = g * y[base + c] + (1.0f - g) * xo[j];
        ss += v[j] * v[j];
    }
    ss = wave_sum32(ss);
    float rr = rsqrtf(ss * (1.0f / DD) + EPS);
#pragma unroll
    for (int j = 0; j < 4; ++j) {
        int c = lane + 32 * j;
        xout[base + c] = lrelu(xo[j] + scale[c] * v[j] * rr);
    }
}

// FiLM layer 1: hidden[b,j] = gelu(z @ w1 + b1),  z:[B,128], w1:[128,256]
__global__ void film1_kernel(const float* __restrict__ z, const float* __restrict__ w1,
                             const float* __restrict__ b1, float* __restrict__ hidden)
{
    int i = blockIdx.x * blockDim.x + threadIdx.x;    // [0, B*256)
    if (i >= BB * 256) return;
    int b = i >> 8, j = i & 255;
    float a = b1[j];
    const float* zr = z + (long)b * DD;
#pragma unroll 4
    for (int k = 0; k < DD; ++k) a += zr[k] * w1[k * 256 + j];
    hidden[i] = gelu_exact(a);
}

// FiLM layer 2: gb = hidden @ w2 + b2,  hidden:[B,256], w2:[256,256]
__global__ void film2_kernel(const float* __restrict__ hidden, const float* __restrict__ w2,
                             const float* __restrict__ b2, float* __restrict__ gb)
{
    int i = blockIdx.x * blockDim.x + threadIdx.x;
    if (i >= BB * 256) return;
    int b = i >> 8, j = i & 255;
    float a = b2[j];
    const float* hr = hidden + (long)b * 256;
#pragma unroll 4
    for (int k = 0; k < 256; ++k) a += hr[k] * w2[k * 256 + j];
    gb[i] = a;
}

// out = (1 + 0.1*tanh(gamma))*x3 + 0.1*tanh(beta), per-graph (256 nodes/graph)
__global__ void film_apply(const float* __restrict__ x3, const float* __restrict__ gb,
                           float* __restrict__ out)
{
    long i = (long)blockIdx.x * blockDim.x + threadIdx.x;
    if (i >= (long)NN * DD) return;
    long n = i >> 7; int d = (int)(i & 127);
    int g = (int)(n >> 8);   // 256 nodes per graph
    float ga = 0.1f * tanhf(gb[g * 256 + d]);
    float be = 0.1f * tanhf(gb[g * 256 + DD + d]);
    out[i] = (1.0f + ga) * x3[i] + be;
}

// ---------------- host orchestration ----------------
extern "C" void kernel_launch(void* const* d_in, const int* in_sizes, int n_in,
                              void* d_out, int out_size, void* d_ws, size_t ws_size,
                              hipStream_t stream)
{
    (void)in_sizes; (void)n_in; (void)out_size; (void)ws_size;

    const float* x_cell     = (const float*)d_in[0];
    const float* x_emb      = (const float*)d_in[1];
    const float* z_h        = (const float*)d_in[2];
    const int*   EI         = (const int*)  d_in[3];
    const float* gat_w[2]   = { (const float*)d_in[4],  (const float*)d_in[8]  };
    const float* gat_as[2]  = { (const float*)d_in[5],  (const float*)d_in[9]  };
    const float* gat_ad[2]  = { (const float*)d_in[6],  (const float*)d_in[10] };
    const float* gat_b[2]   = { (const float*)d_in[7],  (const float*)d_in[11] };
    const float* norm_s[3]  = { (const float*)d_in[12], (const float*)d_in[13], (const float*)d_in[14] };
    const float* kw = (const float*)d_in[15]; const float* kb = (const float*)d_in[16];
    const float* qw = (const float*)d_in[17]; const float* qb = (const float*)d_in[18];
    const float* vw = (const float*)d_in[19]; const float* vb = (const float*)d_in[20];
    const float* rel_k = (const float*)d_in[21];
    const float* rel_v = (const float*)d_in[22];
    const float* rel_p = (const float*)d_in[23];
    const float* aw = (const float*)d_in[24]; const float* ab = (const float*)d_in[25];
    const float* skip = (const float*)d_in[26];
    const float* inj_w = (const float*)d_in[27]; const float* inj_b = (const float*)d_in[28];
    const float* fw1 = (const float*)d_in[29]; const float* fb1 = (const float*)d_in[30];
    const float* fw2 = (const float*)d_in[31]; const float* fb2 = (const float*)d_in[32];

    // workspace layout
    const long ND = (long)NN * DD;           // 16,777,216 floats
    const long NH = (long)NN * HH;
    float* B0   = (float*)d_ws;              // scratch: h_t / kt / vt / y
    float* B1   = B0 + ND;                   // aggregation accumulator
    float* XC   = B1 + ND;                   // x chain (x1 -> x2 -> x3)
    float* Kb   = XC + ND;
    float* Qb   = Kb + ND;
    float* Vb   = Qb + ND;
    float* asrc = Vb + ND;                   // N*H
    float* adst = asrc + NH;
    unsigned* menc = (unsigned*)(adst + NH); // N*H
    float* ssum = (float*)(menc + NH);       // N*H
    float* score = ssum + NH;                // T*E*H
    float* hidden = score + (long)TT * EE * HH;  // B*256
    float* gbuf = hidden + (long)BB * 256;       // B*256

    const dim3 blk(256);
    const int gemmGrid = NN / 128;           // 1024
    const int nhGrid   = (int)((NH + 255) / 256);
    const int fillNDg  = (int)((ND + 255) / 256);
    const int eThreads = (EE * HH + 255) / 256;      // per-(edge,head)
    const int nThreads = (NN * HH + 255) / 256;
    const int eWaves   = EE / 8;                     // wave-per-edge
    const int nWaves   = NN / 8;

    // ---------- two GAT layers ----------
    for (int layer = 0; layer < 2; ++layer) {
        const float* xin = (layer == 0) ? x_cell : XC;
        fill_u32<<<fillNDg, blk, 0, stream>>>((unsigned*)B1, 0u, ND);   // acc = 0
        for (int t = 0; t < TT; ++t) {
            const int* src = EI + (long)t * 2 * EE;
            const int* dst = src + EE;
            // h_t = xin @ W_t
            gemm128_wmma<<<gemmGrid, blk, GEMM_LDS_BYTES, stream>>>(
                xin, gat_w[layer] + (long)t * DD * DD, nullptr, B0, 0, 0, 0);
            gat_attn_coef<<<nhGrid, blk, 0, stream>>>(
                B0, gat_as[layer] + t * HH * HDIM, gat_ad[layer] + t * HH * HDIM, asrc, adst);
            // segment softmax over this type's edges (+ self loops)
            fill_u32<<<nhGrid, blk, 0, stream>>>(menc, 0u, NH);
            gat_edge_max<<<eThreads, blk, 0, stream>>>(src, dst, asrc, adst, menc, EE);
            gat_edge_max<<<nThreads, blk, 0, stream>>>(nullptr, nullptr, asrc, adst, menc, NN);
            fill_u32<<<nhGrid, blk, 0, stream>>>((unsigned*)ssum, 0u, NH);
            gat_edge_sum<<<eThreads, blk, 0, stream>>>(src, dst, asrc, adst, menc, ssum, EE);
            gat_edge_sum<<<nThreads, blk, 0, stream>>>(nullptr, nullptr, asrc, adst, menc, ssum, NN);
            gat_edge_agg<<<eWaves, blk, 0, stream>>>(src, dst, asrc, adst, menc, ssum, B0, B1, EE);
            gat_edge_agg<<<nWaves, blk, 0, stream>>>(nullptr, nullptr, asrc, adst, menc, ssum, B0, B1, NN);
        }
        res_kernel<<<nWaves, blk, 0, stream>>>(xin, B1, gat_b[layer], norm_s[layer], XC);
    }

    // ---------- HGT ----------
    gemm128_wmma<<<gemmGrid, blk, GEMM_LDS_BYTES, stream>>>(XC, kw, kb, Kb, 0, 0, 0);
    gemm128_wmma<<<gemmGrid, blk, GEMM_LDS_BYTES, stream>>>(XC, qw, qb, Qb, 0, 0, 0);
    gemm128_wmma<<<gemmGrid, blk, GEMM_LDS_BYTES, stream>>>(XC, vw, vb, Vb, 0, 0, 0);

    fill_u32<<<nhGrid, blk, 0, stream>>>(menc, 0u, NH);
    for (int t = 0; t < TT; ++t) {
        const int* src = EI + (long)t * 2 * EE;
        const int* dst = src + EE;
        // kt = k @ blockdiag(rel_k[t])
        gemm128_wmma<<<gemmGrid, blk, GEMM_LDS_BYTES, stream>>>(
            Kb, rel_k + (long)t * HH * HDIM * HDIM, nullptr, B0, 1, 0, 0);
        hgt_score_max<<<eWaves, blk, 0, stream>>>(
            src, dst, Qb, B0, rel_p + t * HH, score + (long)t * EE * HH, menc);
    }
    fill_u32<<<nhGrid, blk, 0, stream>>>((unsigned*)ssum, 0u, NH);
    {
        long tot = (long)TT * EE * HH;
        hgt_sum<<<(int)((tot + 255) / 256), blk, 0, stream>>>(score, EI, menc, ssum);
    }
    fill_u32<<<fillNDg, blk, 0, stream>>>((unsigned*)B1, 0u, ND);
    for (int t = 0; t < TT; ++t) {
        const int* src = EI + (long)t * 2 * EE;
        const int* dst = src + EE;
        gemm128_wmma<<<gemmGrid, blk, GEMM_LDS_BYTES, stream>>>(
            Vb, rel_v + (long)t * HH * HDIM * HDIM, nullptr, B0, 1, 0, 0);
        hgt_agg<<<eWaves, blk, 0, stream>>>(
            src, dst, score + (long)t * EE * HH, menc, ssum, B0, B1);
    }
    // y = gelu(acc) @ a_w + a_b
    gemm128_wmma<<<gemmGrid, blk, GEMM_LDS_BYTES, stream>>>(B1, aw, ab, B0, 0, 1, 0);
    hgt_res<<<nWaves, blk, 0, stream>>>(XC, B0, skip, norm_s[2], XC);

    // ---------- input injection: x3 += x_emb @ inj_w + inj_b ----------
    gemm128_wmma<<<gemmGrid, blk, GEMM_LDS_BYTES, stream>>>(x_emb, inj_w, inj_b, XC, 0, 0, 1);

    // ---------- FiLM ----------
    film1_kernel<<<(BB * 256) / 256, blk, 0, stream>>>(z_h, fw1, fb1, hidden);
    film2_kernel<<<(BB * 256) / 256, blk, 0, stream>>>(hidden, fw2, fb2, gbuf);
    film_apply<<<(int)(((long)NN * DD + 255) / 256), blk, 0, stream>>>(XC, gbuf, (float*)d_out);
}
